// ResidualVectorQuantizer_13967233646720
// MI455X (gfx1250) — compile-verified, matrix operand order checked by reading, not run on hardware
//
#include <hip/hip_runtime.h>
#include <hip/hip_bf16.h>
#include <math.h>

#define NQ        4
#define NCODES    1024
#define EMB       128
#define NTOK      131072

typedef __attribute__((ext_vector_type(16))) _Float16 v16h;
typedef __attribute__((ext_vector_type(8)))  _Float16 v8h;
typedef __attribute__((ext_vector_type(8)))  float    v8f;

// ---------------- workspace layout (bytes) ----------------
static constexpr size_t OFF_RESID = 0;                                         // f32 [NTOK*EMB]
static constexpr size_t OFF_CBH   = OFF_RESID + (size_t)NTOK * EMB * 4;        // f16 [NQ*NCODES*EMB]
static constexpr size_t OFF_CN2   = OFF_CBH   + (size_t)NQ * NCODES * EMB * 2; // f32 [NQ*NCODES]
static constexpr size_t OFF_IDX   = OFF_CN2   + (size_t)NQ * NCODES * 4;       // u32 [NTOK]
static constexpr size_t OFF_CNT   = OFF_IDX   + (size_t)NTOK * 4;              // u32 [NQ*NCODES]
static constexpr size_t OFF_COM   = OFF_CNT   + (size_t)NQ * NCODES * 4;       // f32 [NTOK] per-token commit SS
static constexpr size_t OFF_PART  = OFF_COM   + (size_t)NTOK * 4;              // f32 [512] partial sums

// ---------------- init: zero histograms ----------------
__global__ void rvq_init_k(unsigned* __restrict__ counts) {
    int t = blockIdx.x * blockDim.x + threadIdx.x;
    if (t < NQ * NCODES) counts[t] = 0u;
}

// ---------------- codebook prep: f32 -> f16 copy + ||c||^2 ----------------
__global__ void rvq_prep_cb_k(const float* __restrict__ cb,
                              _Float16* __restrict__ cbh,
                              float* __restrict__ cn2) {
    int j = blockIdx.x * blockDim.x + threadIdx.x;       // 0 .. NQ*NCODES-1
    if (j >= NQ * NCODES) return;
    const float* src = cb + (size_t)j * EMB;
    _Float16*    dst = cbh + (size_t)j * EMB;
    float s = 0.0f;
    #pragma unroll 8
    for (int k = 0; k < EMB; ++k) {
        float v = src[k];
        s += v * v;
        dst[k] = (_Float16)v;
    }
    cn2[j] = s;
}

// ---------------- copy z -> resid (stage-0 residual) ----------------
__global__ void rvq_copy_z_k(const float4* __restrict__ z, float4* __restrict__ resid) {
    size_t i = (size_t)blockIdx.x * blockDim.x + threadIdx.x;  // grid sized exactly
    resid[i] = z[i];
}

// ---------------- WMMA argmin: 32 tokens per wave vs 1024 codes ----------------
// Codebook staged through 64KB LDS in 4 chunks of 256 codes, shared by all 8
// waves of the block. Each 256B code row is split into 16x16B parts stored at
// part index (p ^ (code & 15)) -> lanes 0..15 (distinct code&15) read disjoint
// 4-bank groups: conflict-free ds_load_b128.
//
// A fragment (16-bit, 16x32): lane = (M = lane&15, K-half = lane>>4);
//   halfs 0..7  <-> K = 32c + 8*kh + i ; halfs 8..15 <-> K = 32c + 16 + 8*kh + i
// B fragment mirrors A with N in place of M. C fragment (f32 16x16):
//   lane N = lane&15, VGPR r -> M = r + 8*(lane>>4).
__global__ __launch_bounds__(256) void rvq_argmin_k(
    const float*    __restrict__ resid,   // f32 [NTOK, EMB]
    const _Float16* __restrict__ cbh,     // f16 [NCODES, EMB] (stage slice)
    const float*    __restrict__ cn2,     // f32 [NCODES]      (stage slice)
    unsigned*       __restrict__ idxbuf,  // u32 [NTOK]
    unsigned*       __restrict__ counts)  // u32 [NCODES]      (stage slice)
{
    __shared__ uint4 lcb[256 * 16];       // exactly 64 KB

    const int lane = threadIdx.x & 31;
    const int wave = (int)((blockIdx.x * blockDim.x + threadIdx.x) >> 5);
    const int t0   = wave * 32;            // this wave owns tokens [t0, t0+32)
    const int mrow = lane & 15;
    const int kh   = lane >> 4;

    // Load + convert A fragments for both 16-token tiles, all 4 K-chunks.
    v16h afrag[2][4];
    #pragma unroll
    for (int s = 0; s < 2; ++s) {
        const float* arow = resid + (size_t)(t0 + 16 * s + mrow) * EMB;
        #pragma unroll
        for (int c = 0; c < 4; ++c) {
            const int base = 32 * c + 8 * kh;
            #pragma unroll
            for (int i = 0; i < 8; ++i) {
                afrag[s][c][i]     = (_Float16)arow[base + i];
                afrag[s][c][8 + i] = (_Float16)arow[base + 16 + i];
            }
        }
    }

    float    bestv[2][8];
    unsigned besti[2][8];
    #pragma unroll
    for (int s = 0; s < 2; ++s)
        #pragma unroll
        for (int r = 0; r < 8; ++r) { bestv[s][r] = 3.0e38f; besti[s][r] = 0u; }

    for (int ch = 0; ch < 4; ++ch) {       // 256 codes per LDS chunk
        __syncthreads();
        {   // cooperative staging: 4096 uint4 = 64KB, 16 per thread, XOR-swizzled
            const uint4* src = (const uint4*)cbh + (size_t)ch * 4096;
            #pragma unroll
            for (int it = 0; it < 16; ++it) {
                const int i = threadIdx.x + it * 256;
                const int j = i >> 4;       // code within chunk
                const int p = i & 15;       // 16B part within row
                lcb[j * 16 + (p ^ (j & 15))] = src[i];
            }
        }
        __syncthreads();

        for (int ntl = 0; ntl < 16; ++ntl) {
            const int nt = ch * 16 + ntl;
            const unsigned code = (unsigned)(nt * 16 + mrow);   // this lane's N column
            const v8h* brow = (const v8h*)(lcb + (size_t)(ntl * 16 + mrow) * 16);
            v8f acc0 = {}, acc1 = {};
            #pragma unroll
            for (int c = 0; c < 4; ++c) {
                const v8h g1 = brow[(4 * c + kh)     ^ mrow];   // K = 32c+8kh   .. +7
                const v8h g2 = brow[(4 * c + kh + 2) ^ mrow];   // K = 32c+16+8kh.. +7
                const v16h bfrag = __builtin_shufflevector(
                    g1, g2, 0, 1, 2, 3, 4, 5, 6, 7, 8, 9, 10, 11, 12, 13, 14, 15);
                acc0 = __builtin_amdgcn_wmma_f32_16x16x32_f16(
                           false, afrag[0][c], false, bfrag, (short)0, acc0, false, false);
                acc1 = __builtin_amdgcn_wmma_f32_16x16x32_f16(
                           false, afrag[1][c], false, bfrag, (short)0, acc1, false, false);
            }
            const float cn = cn2[code];
            #pragma unroll
            for (int r = 0; r < 8; ++r) {
                float d0 = cn - 2.0f * acc0[r];
                if (d0 < bestv[0][r]) { bestv[0][r] = d0; besti[0][r] = code; }
                float d1 = cn - 2.0f * acc1[r];
                if (d1 < bestv[1][r]) { bestv[1][r] = d1; besti[1][r] = code; }
            }
        }
    }

    // Min-reduce (value, then lower index) across the 16 lanes of each half-wave.
    #pragma unroll
    for (int s = 0; s < 2; ++s) {
        #pragma unroll
        for (int r = 0; r < 8; ++r) {
            float    v = bestv[s][r];
            unsigned i = besti[s][r];
            #pragma unroll
            for (int off = 8; off >= 1; off >>= 1) {
                float    ov = __shfl_xor(v, off, 32);
                unsigned oi = (unsigned)__shfl_xor((int)i, off, 32);
                if (ov < v || (ov == v && oi < i)) { v = ov; i = oi; }
            }
            if ((lane & 15) == 0) {
                const int tok = t0 + 16 * s + r + 8 * kh;
                idxbuf[tok] = i;
                atomicAdd(&counts[i], 1u);   // integer: exactly commutative
            }
        }
    }
}

// ---------------- residual update / gather / commit / idx output ----------------
__global__ __launch_bounds__(256) void rvq_update_k(
    float*          __restrict__ resid,     // f32 [NTOK, EMB] (in/out)
    const float*    __restrict__ cbq,       // f32 [NCODES, EMB] original codebook slice
    const unsigned* __restrict__ idxbuf,    // u32 [NTOK]
    float*          __restrict__ zqsum,     // f32 [NTOK, EMB]  (d_out head)
    float*          __restrict__ outidx,    // f32 [NTOK]       (d_out idx slice for stage)
    float*          __restrict__ commitbuf, // f32 [NTOK] per-token SS accumulator
    int isFirst)
{
    const int lane = threadIdx.x & 31;
    const int tok  = (int)((blockIdx.x * blockDim.x + threadIdx.x) >> 5);
    const unsigned code = idxbuf[tok];

    const float4* c4 = (const float4*)(cbq + (size_t)code * EMB) + lane;
    float4*       r4 = (float4*)(resid + (size_t)tok * EMB) + lane;
    float4*       o4 = (float4*)(zqsum + (size_t)tok * EMB) + lane;

    float4 c = *c4;
    float4 r = *r4;
    float4 nr = make_float4(r.x - c.x, r.y - c.y, r.z - c.z, r.w - c.w);
    float ss = nr.x * nr.x + nr.y * nr.y + nr.z * nr.z + nr.w * nr.w; // (zq - z)^2
    *r4 = nr;
    if (isFirst) {
        *o4 = c;
    } else {
        float4 o = *o4;
        o.x += c.x; o.y += c.y; o.z += c.z; o.w += c.w;
        *o4 = o;
    }
    #pragma unroll
    for (int off = 16; off >= 1; off >>= 1) ss += __shfl_xor(ss, off, 32);
    if (lane == 0) {
        if (isFirst) commitbuf[tok] = ss;
        else         commitbuf[tok] += ss;      // fixed-order: deterministic
        outidx[tok] = (float)code;
    }
}

// ---------------- deterministic commit reduction: NTOK -> 512 partials ----------------
__global__ __launch_bounds__(256) void rvq_commit_reduce_k(
    const float* __restrict__ commitbuf, float* __restrict__ partials)
{
    __shared__ float red[256];
    const int t = threadIdx.x;
    red[t] = commitbuf[(size_t)blockIdx.x * 256 + t];
    __syncthreads();
    for (int s = 128; s > 0; s >>= 1) {
        if (t < s) red[t] += red[t + s];
        __syncthreads();
    }
    if (t == 0) partials[blockIdx.x] = red[0];
}

// ---------------- perplexity + commit scalars ----------------
__global__ __launch_bounds__(1024) void rvq_finalize_k(
    const unsigned* __restrict__ counts,   // u32 [NQ*NCODES]
    const float*    __restrict__ partials, // f32 [512]
    float*          __restrict__ out)      // -> d_out + NTOK*EMB : [commit, perp]
{
    __shared__ float red[1024];
    const int t = threadIdx.x;

    // deterministic sum of the 512 commit partials
    red[t] = (t < 512) ? partials[t] : 0.0f;
    __syncthreads();
    for (int s = 512; s > 0; s >>= 1) {
        if (t < s) red[t] += red[t + s];
        __syncthreads();
    }
    const float commitTotal = red[0];
    __syncthreads();

    float perp = 0.0f;
    for (int q = 0; q < NQ; ++q) {
        float p = (float)counts[q * NCODES + t] * (1.0f / (float)NTOK);
        red[t] = p * logf(p + 1e-10f);
        __syncthreads();
        for (int s = 512; s > 0; s >>= 1) {
            if (t < s) red[t] += red[t + s];
            __syncthreads();
        }
        if (t == 0) perp += expf(-red[0]);
        __syncthreads();
    }
    if (t == 0) {
        out[0] = commitTotal * 0.25f / ((float)NTOK * (float)EMB * (float)NQ);
        out[1] = perp / (float)NQ;
    }
}

// ---------------- host launch ----------------
extern "C" void kernel_launch(void* const* d_in, const int* in_sizes, int n_in,
                              void* d_out, int out_size, void* d_ws, size_t ws_size,
                              hipStream_t stream) {
    (void)in_sizes; (void)n_in; (void)out_size; (void)ws_size;
    const float* z  = (const float*)d_in[0];   // [NTOK, EMB]
    const float* cb = (const float*)d_in[1];   // [NQ, NCODES, EMB]
    float* out = (float*)d_out;

    char* ws = (char*)d_ws;
    float*     resid     = (float*)(ws + OFF_RESID);
    _Float16*  cbh       = (_Float16*)(ws + OFF_CBH);
    float*     cn2       = (float*)(ws + OFF_CN2);
    unsigned*  idxbuf    = (unsigned*)(ws + OFF_IDX);
    unsigned*  counts    = (unsigned*)(ws + OFF_CNT);
    float*     commitbuf = (float*)(ws + OFF_COM);
    float*     partials  = (float*)(ws + OFF_PART);

    const size_t NZ = (size_t)NTOK * EMB;

    rvq_init_k<<<(NQ * NCODES + 255) / 256, 256, 0, stream>>>(counts);
    rvq_prep_cb_k<<<(NQ * NCODES) / 256, 256, 0, stream>>>(cb, cbh, cn2);
    rvq_copy_z_k<<<(int)(NZ / 4 / 256), 256, 0, stream>>>((const float4*)z, (float4*)resid);

    for (int q = 0; q < NQ; ++q) {
        rvq_argmin_k<<<NTOK / 32 / 8, 256, 0, stream>>>(
            resid,
            cbh + (size_t)q * NCODES * EMB,
            cn2 + (size_t)q * NCODES,
            idxbuf,
            counts + (size_t)q * NCODES);
        rvq_update_k<<<NTOK / 8, 256, 0, stream>>>(
            resid,
            cb + (size_t)q * NCODES * EMB,
            idxbuf,
            out,
            out + NZ + 2 + (size_t)q * NTOK,
            commitbuf,
            q == 0 ? 1 : 0);
    }

    rvq_commit_reduce_k<<<NTOK / 256, 256, 0, stream>>>(commitbuf, partials);
    rvq_finalize_k<<<1, 1024, 0, stream>>>(counts, partials, out + NZ);
}
